// Fuzzy_Attention_Encoder_69200513073620
// MI455X (gfx1250) — compile-verified, matrix operand order checked by reading, not run on hardware
//
#include <hip/hip_runtime.h>

typedef __attribute__((ext_vector_type(16))) _Float16 v16h;
typedef __attribute__((ext_vector_type(8)))  float    v8f;
typedef __attribute__((ext_vector_type(4)))  unsigned u32x4;
typedef __attribute__((ext_vector_type(8)))  unsigned u32x8;

#define BB 64
#define SS 4
#define LL 64
#define DD 64
#define RR 64
#define NROW (BB*SS*LL)      // 16384 rows of [D]
#define F1 16384
#define HH 8192

__device__ __forceinline__ v8f wmma16(v16h a, v16h b, v8f c) {
  // v_wmma_f32_16x16x32_f16, f32 accumulate
  return __builtin_amdgcn_wmma_f32_16x16x32_f16(false, a, false, b, (short)0, c, false, false);
}

// 16x32 f16 operand tile from a row-major f32 matrix (row stride ld).
// ISA 16-bit A layout (B uses the transpose-symmetric pattern from weight rows):
//   lane<16 : row=lane,    halves 0..7 = K 0..7,  halves 8..15 = K 16..23
//   lane>=16: row=lane-16, halves 0..7 = K 8..15, halves 8..15 = K 24..31
__device__ __forceinline__ v16h ld_tile_f32(const float* p, int ld, int lane) {
  int r = lane & 15;
  int k0 = (lane >> 4) << 3;
  const float* row = p + (long)r * ld;
  v16h t;
#pragma unroll
  for (int i = 0; i < 8; ++i) t[i]     = (_Float16)row[k0 + i];
#pragma unroll
  for (int i = 0; i < 8; ++i) t[8 + i] = (_Float16)row[16 + k0 + i];
  return t;
}

__device__ __forceinline__ v16h ld_tile_f32_sq(const float* p, int ld, int lane) {
  int r = lane & 15;
  int k0 = (lane >> 4) << 3;
  const float* row = p + (long)r * ld;
  v16h t;
#pragma unroll
  for (int i = 0; i < 8; ++i) { float v = row[k0 + i];      t[i]     = (_Float16)(v * v); }
#pragma unroll
  for (int i = 0; i < 8; ++i) { float v = row[16 + k0 + i]; t[8 + i] = (_Float16)(v * v); }
  return t;
}

__device__ __forceinline__ v16h ld_tile_f16(const _Float16* p, int ld, int lane) {
  int r = lane & 15;
  int k0 = (lane >> 4) << 3;
  const _Float16* row = p + (long)r * ld;
  v16h t;
#pragma unroll
  for (int i = 0; i < 8; ++i) t[i]     = row[k0 + i];
#pragma unroll
  for (int i = 0; i < 8; ++i) t[8 + i] = row[16 + k0 + i];
  return t;
}

// P tile: P[m, k] = F[m, r] * X[m, d0 + k_local]   (outer-product A operand)
__device__ __forceinline__ v16h ld_tile_outer(const float* xs, const float* fs,
                                              int d0, int r, int lane) {
  int m  = lane & 15;
  int k0 = (lane >> 4) << 3;
  float f = fs[m * 64 + r];
  const float* xr = xs + m * 64 + d0;
  v16h t;
#pragma unroll
  for (int i = 0; i < 8; ++i) t[i]     = (_Float16)(f * xr[k0 + i]);
#pragma unroll
  for (int i = 0; i < 8; ++i) t[8 + i] = (_Float16)(f * xr[16 + k0 + i]);
  return t;
}

// ---- CDNA5 async global->LDS copy (per-lane b128), tracked by ASYNCcnt
__device__ __forceinline__ void async_copy16(const float* g, unsigned lds_off) {
  asm volatile("global_load_async_to_lds_b128 %0, %1, off"
               :: "v"(lds_off), "v"(g) : "memory");
}
__device__ __forceinline__ void wait_async0() {
  asm volatile("s_wait_asynccnt 0x0" ::: "memory");
}
__device__ __forceinline__ void wait_ds0() {
  asm volatile("s_wait_dscnt 0x0" ::: "memory");
}

// ---- CDNA5 TDM: DMA a 2D fp32 tile (32 rows x 32 cols, row stride = ldElems)
// into LDS at lds_addr. D# per CDNA5 ISA (2-group form,
// VADDR2/3 = NULL -> up to 2D tensor). Tracked by TENSORcnt.
__device__ __forceinline__ void tdm_load_tile32x32(const float* gsrc, unsigned lds_addr,
                                                   unsigned ldElems, unsigned nrows) {
  unsigned long long g = (unsigned long long)(const void*)gsrc;
  u32x4 g0;
  g0[0] = 1u;                                            // count=1, user mode
  g0[1] = lds_addr;                                      // lds_addr [63:32]
  g0[2] = (unsigned)g;                                   // global_addr lo
  g0[3] = (unsigned)((g >> 32) & 0x01FFFFFFull) | (2u << 30); // addr hi | type=2
  u32x8 g1;
  g1[0] = 0x00020000u;                                   // data_size=2 (4B)
  g1[1] = (ldElems & 0xFFFFu) << 16;                     // tensor_dim0 [15:0]
  g1[2] = ((ldElems >> 16) & 0xFFFFu)                    // tensor_dim0 [31:16]
        | ((nrows & 0xFFFFu) << 16);                     // tensor_dim1 [15:0]
  g1[3] = ((nrows >> 16) & 0xFFFFu) | (32u << 16);       // dim1 hi | tile_dim0=32
  g1[4] = 32u;                                           // tile_dim1=32, tile_dim2=0
  g1[5] = ldElems;                                       // tensor_dim0_stride lo32
  g1[6] = 0u;                                            // stride hi | dim1_stride lo
  g1[7] = 0u;
  asm volatile("tensor_load_to_lds %0, %1" :: "s"(g0), "s"(g1) : "memory");
}

// ---------------- K0: rule coefficients  inv=1/w^2, nb=-2c/w^2, k_r=sum c^2/w^2
__global__ void k_rulecoef(const float* __restrict__ centers, const float* __restrict__ widths,
                           _Float16* invh, _Float16* nbh, float* krule) {
  __shared__ float red[64];
  int r = blockIdx.x;       // R=64
  int d = threadIdx.x;      // D=64
  float w = widths[r * 64 + d];
  float c = centers[r * 64 + d];
  float iv = 1.0f / (w * w);
  invh[r * 64 + d] = (_Float16)iv;
  nbh[r * 64 + d]  = (_Float16)(-2.0f * c * iv);
  red[d] = c * c * iv;
  __syncthreads();
#pragma unroll
  for (int s = 32; s > 0; s >>= 1) { if (d < s) red[d] += red[d + s]; __syncthreads(); }
  if (d == 0) krule[r] = red[0];
}

// ---------------- K1: q = x@Wq^T+bq ; z = -(1/128)*(q^2@inv^T + q@nb^T + k) ; softmax -> Fss
__global__ void __launch_bounds__(32)
k_fuzzy(const float* __restrict__ x, const float* __restrict__ Wq, const float* __restrict__ bq,
        const _Float16* __restrict__ invh, const _Float16* __restrict__ nbh,
        const float* __restrict__ krule, float* __restrict__ Fss) {
  __shared__ float qs[16 * 64];
  __shared__ float zs[16 * 64];
  int lane = threadIdx.x;
  long row0 = (long)blockIdx.x * 16;
  const float* xrow = x + row0 * 64;

  // Q projection: 4 n-tiles x 2 k-chunks
#pragma unroll
  for (int nt = 0; nt < 4; ++nt) {
    v8f acc = {};
#pragma unroll
    for (int kc = 0; kc < 64; kc += 32) {
      v16h a = ld_tile_f32(xrow + kc, 64, lane);
      v16h b = ld_tile_f32(Wq + (long)(nt * 16) * 64 + kc, 64, lane);
      acc = wmma16(a, b, acc);
    }
    int col = nt * 16 + (lane & 15);
    int mh  = (lane >> 4) * 8;
    float bqv = bq[col];
#pragma unroll
    for (int v = 0; v < 8; ++v) qs[(v + mh) * 64 + col] = acc[v] + bqv;
  }
  __syncthreads();

  // z = -(0.5/64) * (Q^2 @ inv^T + Q @ (-2 c inv)^T + k_r)
#pragma unroll
  for (int nt = 0; nt < 4; ++nt) {
    v8f acc = {};
#pragma unroll
    for (int kc = 0; kc < 64; kc += 32) {
      v16h a2 = ld_tile_f32_sq(qs + kc, 64, lane);
      v16h bi = ld_tile_f16(invh + (long)(nt * 16) * 64 + kc, 64, lane);
      acc = wmma16(a2, bi, acc);
      v16h a1 = ld_tile_f32(qs + kc, 64, lane);
      v16h bn = ld_tile_f16(nbh + (long)(nt * 16) * 64 + kc, 64, lane);
      acc = wmma16(a1, bn, acc);
    }
    int col = nt * 16 + (lane & 15);
    int mh  = (lane >> 4) * 8;
    float kr = krule[col];
#pragma unroll
    for (int v = 0; v < 8; ++v) zs[(v + mh) * 64 + col] = -0.0078125f * (acc[v] + kr);
  }
  __syncthreads();

  // softmax over 64 rules, one row at a time (lane holds cols lane, lane+32)
  for (int m = 0; m < 16; ++m) {
    float v0 = zs[m * 64 + lane];
    float v1 = zs[m * 64 + 32 + lane];
    float mx = fmaxf(v0, v1);
#pragma unroll
    for (int off = 16; off > 0; off >>= 1) mx = fmaxf(mx, __shfl_xor(mx, off, 32));
    float e0 = __expf(v0 - mx), e1 = __expf(v1 - mx);
    float s = e0 + e1;
#pragma unroll
    for (int off = 16; off > 0; off >>= 1) s += __shfl_xor(s, off, 32);
    float inv = 1.0f / s;
    Fss[(row0 + m) * 64 + lane]      = e0 * inv;
    Fss[(row0 + m) * 64 + 32 + lane] = e1 * inv;
  }
}

// ---------------- K2: attn = (F (x) x) @ Wc2^T + F@Bc^T ; h = x+attn ; batch LN stats
__global__ void __launch_bounds__(32)
k_attn(const float* __restrict__ x, const float* __restrict__ Wc, const float* __restrict__ bc,
       const float* __restrict__ Fss, float* __restrict__ h,
       float* bsum, float* bsumsq) {
  __shared__ float xsm[16 * 64];
  __shared__ float fsm[16 * 64];
  int lane = threadIdx.x;
  long row0 = (long)blockIdx.x * 16;

  // stage x and F tiles via CDNA5 async global->LDS copies (ASYNCcnt)
  {
    const float* xg = x   + row0 * 64;
    const float* fg = Fss + row0 * 64;
    unsigned xl = (unsigned)(size_t)(void*)xsm;
    unsigned fl = (unsigned)(size_t)(void*)fsm;
#pragma unroll
    for (int j = 0; j < 8; ++j) {
      int i = j * 32 + lane;                 // 256 x 16B per array
      async_copy16(xg + i * 4, xl + i * 16);
      async_copy16(fg + i * 4, fl + i * 16);
    }
    wait_async0();
  }
  __syncthreads();

  v8f acc[4] = {v8f{}, v8f{}, v8f{}, v8f{}};
  // K = 4096 = r(64) x d(64); per chunk of 32 the rule index r is fixed
  for (int r = 0; r < RR; ++r) {
#pragma unroll
    for (int kc = 0; kc < 64; kc += 32) {
      v16h a = ld_tile_outer(xsm, fsm, kc, r, lane);
#pragma unroll
      for (int nt = 0; nt < 4; ++nt) {
        // B col t -> Wc row (t*64 + r), cols kc..kc+31 ; t-stride = 64*64
        v16h b = ld_tile_f32(Wc + ((long)(nt * 16) * 64 + r) * 64 + kc, 64 * 64, lane);
        acc[nt] = wmma16(a, b, acc[nt]);
      }
    }
  }
  // bias term: attn += F @ Bc^T,  Bc[t,r] = bc[t*64+r]
#pragma unroll
  for (int kc = 0; kc < 64; kc += 32) {
    v16h a = ld_tile_f32(fsm + kc, 64, lane);
#pragma unroll
    for (int nt = 0; nt < 4; ++nt) {
      v16h b = ld_tile_f32(bc + (long)(nt * 16) * 64 + kc, 64, lane);
      acc[nt] = wmma16(a, b, acc[nt]);
    }
  }

  float s1 = 0.f, s2 = 0.f;
  int colb = lane & 15, mh = (lane >> 4) * 8;
#pragma unroll
  for (int nt = 0; nt < 4; ++nt) {
    int col = nt * 16 + colb;
#pragma unroll
    for (int v = 0; v < 8; ++v) {
      int m = v + mh;
      float hv = xsm[m * 64 + col] + acc[nt][v];
      h[(row0 + m) * 64 + col] = hv;
      s1 += hv; s2 += hv * hv;
    }
  }
#pragma unroll
  for (int off = 16; off > 0; off >>= 1) {
    s1 += __shfl_xor(s1, off, 32);
    s2 += __shfl_xor(s2, off, 32);
  }
  if (lane == 0) {
    int b = (int)(row0 >> 8);           // 256 rows per batch element
    atomicAdd(&bsum[b], s1);
    atomicAdd(&bsumsq[b], s2);
  }
}

// ---------------- K3: LayerNorm over (S,L,D) per batch
__global__ void k_ln(const float* __restrict__ h, const float* __restrict__ bsum,
                     const float* __restrict__ bsumsq, const float* __restrict__ g,
                     const float* __restrict__ bta, float* __restrict__ hn) {
  long idx = (long)blockIdx.x * 256 + threadIdx.x;   // 64*16384 elems
  int b = (int)(idx >> 14);
  int k = (int)(idx & 16383);
  float mu  = bsum[b]   * (1.0f / 16384.0f);
  float var = bsumsq[b] * (1.0f / 16384.0f) - mu * mu;
  float rs  = rsqrtf(var + 1e-5f);
  hn[idx] = (h[idx] - mu) * rs * g[k] + bta[k];
}

// ---------------- K4: o1 = relu(hn[64,16384] @ W1^T[16384,8192] + b1)
// HBM-bound on W1 (512 MB fp32). W1 tiles streamed by the Tensor Data Mover
// into double-buffered LDS (TENSORcnt), WMMA consumes from LDS with in-register
// f32->f16 conversion; A (hn, 4MB) stays L2-resident via direct global loads.
__global__ void __launch_bounds__(32)
k_ffn1(const float* __restrict__ hn, const float* __restrict__ W1,
       const float* __restrict__ b1, float* __restrict__ o1) {
  __shared__ float wbuf[2][32 * 32];                 // 2 x 4KB W1 tile
  int lane = threadIdx.x;
  int n0 = blockIdx.x * 32;                          // 256 blocks x 32 cols
  unsigned lds0 = (unsigned)(size_t)(void*)&wbuf[0][0];
  unsigned lds1 = (unsigned)(size_t)(void*)&wbuf[1][0];
  const float* wslab = W1 + (long)n0 * F1;           // 32 rows of W1

  v8f acc[4][2] = {{v8f{}, v8f{}}, {v8f{}, v8f{}}, {v8f{}, v8f{}}, {v8f{}, v8f{}}};

  const int NIT = F1 / 32;                           // 512 k-steps
  tdm_load_tile32x32(wslab, lds0, F1, 32);           // prologue DMA
  for (int it = 0; it < NIT; ++it) {
    int kc = it * 32;
    if (it + 1 < NIT) {
      tdm_load_tile32x32(wslab + kc + 32, (it & 1) ? lds0 : lds1, F1, 32);
      __builtin_amdgcn_s_wait_tensorcnt(1);          // current buffer ready
    } else {
      __builtin_amdgcn_s_wait_tensorcnt(0);
    }
    const float* wb = wbuf[it & 1];
    v16h a[4];
#pragma unroll
    for (int mt = 0; mt < 4; ++mt)
      a[mt] = ld_tile_f32(hn + (long)(mt * 16) * F1 + kc, F1, lane);
#pragma unroll
    for (int nt = 0; nt < 2; ++nt) {
      v16h bt = ld_tile_f32(wb + (nt * 16) * 32, 32, lane);
#pragma unroll
      for (int mt = 0; mt < 4; ++mt) acc[mt][nt] = wmma16(a[mt], bt, acc[mt][nt]);
    }
    wait_ds0();   // LDS reads retired before TDM may overwrite this buffer
  }

  int colb = lane & 15, mh = (lane >> 4) * 8;
#pragma unroll
  for (int mt = 0; mt < 4; ++mt)
#pragma unroll
    for (int nt = 0; nt < 2; ++nt) {
      int col = n0 + nt * 16 + colb;
      float bv = b1[col];
#pragma unroll
      for (int v = 0; v < 8; ++v) {
        int brow = mt * 16 + v + mh;
        o1[(long)brow * HH + col] = fmaxf(acc[mt][nt][v] + bv, 0.0f);
      }
    }
}

// ---------------- K5: o2 = relu(o1[64,8192] @ W2^T[8192,64] + b2)
__global__ void __launch_bounds__(32)
k_ffn2(const float* __restrict__ o1, const float* __restrict__ W2,
       const float* __restrict__ b2, float* __restrict__ o2) {
  int lane = threadIdx.x;
  int n0 = blockIdx.x * 16;                          // 4 blocks x 16 cols
  v8f acc[4] = {v8f{}, v8f{}, v8f{}, v8f{}};
  for (int kc = 0; kc < HH; kc += 32) {
    v16h bt = ld_tile_f32(W2 + (long)n0 * HH + kc, HH, lane);
#pragma unroll
    for (int mt = 0; mt < 4; ++mt) {
      v16h a = ld_tile_f32(o1 + (long)(mt * 16) * HH + kc, HH, lane);
      acc[mt] = wmma16(a, bt, acc[mt]);
    }
  }
  int colb = lane & 15, mh = (lane >> 4) * 8;
#pragma unroll
  for (int mt = 0; mt < 4; ++mt) {
    int col = n0 + colb;
    float bv = b2[col];
#pragma unroll
    for (int v = 0; v < 8; ++v)
      o2[(mt * 16 + v + mh) * 64 + col] = fmaxf(acc[mt][v] + bv, 0.0f);
  }
}

// ---------------- K6: out[64,2] = o2 @ W3^T + b3
__global__ void k_head(const float* __restrict__ o2, const float* __restrict__ W3,
                       const float* __restrict__ b3, float* __restrict__ out) {
  int b = threadIdx.x;                               // 64 threads
  float s0 = b3[0], s1 = b3[1];
#pragma unroll
  for (int k = 0; k < 64; ++k) {
    float v = o2[b * 64 + k];
    s0 += v * W3[k];
    s1 += v * W3[64 + k];
  }
  out[b * 2 + 0] = s0;
  out[b * 2 + 1] = s1;
}

extern "C" void kernel_launch(void* const* d_in, const int* in_sizes, int n_in,
                              void* d_out, int out_size, void* d_ws, size_t ws_size,
                              hipStream_t stream) {
  const float* x       = (const float*)d_in[0];
  const float* Wq      = (const float*)d_in[1];
  const float* bq      = (const float*)d_in[2];
  const float* centers = (const float*)d_in[3];
  const float* widths  = (const float*)d_in[4];
  const float* Wc      = (const float*)d_in[5];
  const float* bc      = (const float*)d_in[6];
  const float* ln_g    = (const float*)d_in[7];
  const float* ln_b    = (const float*)d_in[8];
  const float* W1      = (const float*)d_in[9];
  const float* b1      = (const float*)d_in[10];
  const float* W2      = (const float*)d_in[11];
  const float* b2      = (const float*)d_in[12];
  const float* W3      = (const float*)d_in[13];
  const float* b3      = (const float*)d_in[14];
  float* out = (float*)d_out;

  char* ws = (char*)d_ws;
  _Float16* invh  = (_Float16*)(ws + 0);              //  8 KB
  _Float16* nbh   = (_Float16*)(ws + 8192);           //  8 KB
  float*    krule = (float*)  (ws + 16384);           // 256 B
  float*    bsum  = (float*)  (ws + 16640);           // 256 B
  float*    bsumsq= (float*)  (ws + 16896);           // 256 B
  float*    Fss   = (float*)  (ws + 32768);           // 4 MB
  float*    h     = (float*)  (ws + 32768 + (1 << 22));          // 4 MB
  float*    hn    = (float*)  (ws + 32768 + 2 * (1 << 22));      // 4 MB
  float*    o1    = (float*)  (ws + 32768 + 3 * (1 << 22));      // 2 MB
  float*    o2    = (float*)  (ws + 32768 + 3 * (1 << 22) + (1 << 21));

  hipMemsetAsync(bsum, 0, 512, stream);   // zeros bsum + bsumsq

  k_rulecoef<<<dim3(RR), dim3(DD), 0, stream>>>(centers, widths, invh, nbh, krule);
  k_fuzzy   <<<dim3(NROW / 16), dim3(32), 0, stream>>>(x, Wq, bq, invh, nbh, krule, Fss);
  k_attn    <<<dim3(NROW / 16), dim3(32), 0, stream>>>(x, Wc, bc, Fss, h, bsum, bsumsq);
  k_ln      <<<dim3((BB * F1) / 256), dim3(256), 0, stream>>>(h, bsum, bsumsq, ln_g, ln_b, hn);
  k_ffn1    <<<dim3(HH / 32), dim3(32), 0, stream>>>(hn, W1, b1, o1);
  k_ffn2    <<<dim3(4), dim3(32), 0, stream>>>(o1, W2, b2, o2);
  k_head    <<<dim3(1), dim3(64), 0, stream>>>(o2, W3, b3, out);
  (void)in_sizes; (void)n_in; (void)out_size; (void)ws_size;
}
